// cycnet_75376676045162
// MI455X (gfx1250) — matrix operand order repaired and checked by reading, not compiled, not run on hardware
//
#include <hip/hip_runtime.h>

#define NV 255
#define NL 16
#define NE 4080        // NV*NL (real per-slab elements)
#define NEP 4096       // padded slab stride (elements)
#define NP 8
#define PE 32640       // NP*NE (message elements per batch)
#define PEP 32768      // NP*NEP (padded LDS elements)
#define ROWSP 2048     // NP*256 padded rows
#define NTILES 128     // ROWSP/16
#define CLIPV 10.0f

typedef __attribute__((ext_vector_type(16))) _Float16 v16h;
typedef __attribute__((ext_vector_type(8)))  float    v8f;

__launch_bounds__(512)
__global__ void cycnet_bp_kernel(const float* __restrict__ x,
                                 const float* __restrict__ message,
                                 const float* __restrict__ oddw_v,
                                 const float* __restrict__ oddw_e,
                                 const float* __restrict__ w_e_out,
                                 const int*   __restrict__ perma,
                                 const int*   __restrict__ rc_idx,
                                 float*       __restrict__ out)
{
    // Entire per-batch-element BP state lives in LDS (MI455X: 320KB/WGP).
    // Slab p occupies [p*4096, p*4096+4080) real + 16 pad slots; row (p,v) at g*16, g=p*256+v.
    __shared__ _Float16 bufA[PEP];    // ev (even-layer out / odd-layer in), fp16
    __shared__ _Float16 bufB[PEP];    // odd-layer output, scattered via rc
    __shared__ int      rc_s[NEP];    // rc + identity pad entries
    __shared__ float    xv_s[ROWSP];  // x[perma[p, v+1]] per padded row; reused for eo
    __shared__ float    we_s[NL*NL];
    __shared__ float    wv_s[NL];
    __shared__ float    weo_s[NL];

    const unsigned tid = threadIdx.x;
    const unsigned bs  = blockDim.x;
    const unsigned b   = blockIdx.x;
    const float* xr = x       + (size_t)b * (NV + 1);
    const float* mr = message + (size_t)b * PE;

    for (unsigned i = tid; i < NEP; i += bs)
        rc_s[i] = (i < NE) ? rc_idx[i] : (int)i;         // pad -> identity (slab pad slots)
    for (unsigned i = tid; i < PEP; i += bs) {
        unsigned p = i >> 12, j = i & 4095u;
        bufA[i] = (j < NE) ? (_Float16)mr[p * NE + j] : (_Float16)0.0f;
    }
    for (unsigned g = tid; g < ROWSP; g += bs) {
        unsigned p = g >> 8, v = g & 255u;
        xv_s[g] = (v < NV) ? xr[perma[p * (NV + 1) + v + 1]] : 0.0f;
    }
    if (tid < NL) weo_s[tid] = w_e_out[tid];

    const unsigned lane = tid & 31u;
    const unsigned wave = tid >> 5;
    const unsigned nwav = bs >> 5;
    const unsigned chan = lane & 15u;       // A row select (load) / D column n (epilogue)
    const unsigned hi   = (lane >> 4) & 1u; // half-wave select
    const unsigned mb   = hi << 3;          // epilogue row base within tile

    for (int t = 0; t < 5; ++t) {
        if (tid < NL * NL) we_s[tid] = oddw_e[t * NL * NL + tid];
        if (tid < NL)      wv_s[tid] = oddw_v[t * NL + tid];
        __syncthreads();

        // B operand: rows K=0..15 = masked weight W[l,m] (diag zeroed), K=16..31 = 0 pad.
        v16h bm = {};
        {
            #pragma unroll
            for (int n = 0; n < 8; ++n) {
                const unsigned k = (hi << 3) + n;
                float w = we_s[k * 16 + chan];
                bm[n] = (_Float16)((k == chan) ? 0.0f : w);
            }
        }
        const float wvc = wv_s[chan];

        // ---- odd layer: 128 branch-free tiles, one v_wmma_f32_16x16x32_f16 each ----
        for (unsigned T = wave; T < NTILES; T += nwav) {
            const unsigned p     = T >> 4;          // uniform per tile
            const unsigned vbase = (T & 15u) << 4;
            const unsigned slab  = p << 12;

            // A slice: lane<16 -> row v=vbase+chan K=0..7; lane>=16 -> same row K=8..15.
            const int* rcb = rc_s + ((vbase + chan) << 4) + (hi << 3);
            const int4 i0 = *(const int4*)rcb;
            const int4 i1 = *(const int4*)(rcb + 4);
            v16h a = {};                             // K halves 8..15 stay zero (pad)
            a[0] = bufA[slab + i0.x]; a[1] = bufA[slab + i0.y];
            a[2] = bufA[slab + i0.z]; a[3] = bufA[slab + i0.w];
            a[4] = bufA[slab + i1.x]; a[5] = bufA[slab + i1.y];
            a[6] = bufA[slab + i1.z]; a[7] = bufA[slab + i1.w];

            v8f c = {};
            v8f d = __builtin_amdgcn_wmma_f32_16x16x32_f16(
                false, a, false, bm, (short)0, c, false, false);

            // Epilogue: lane holds D[mb..mb+7, chan]; rows vr = vbase+mb+r (same slab).
            const unsigned rbase = (p << 8) + vbase + mb;
            const float4 xa = *(const float4*)(xv_s + rbase);
            const float4 xb = *(const float4*)(xv_s + rbase + 4);
            const float xv8[8] = {xa.x, xa.y, xa.z, xa.w, xb.x, xb.y, xb.z, xb.w};
            #pragma unroll
            for (int r = 0; r < 8; ++r) {
                float z = d[r] + xv8[r] * wvc;
                z = fminf(CLIPV, fmaxf(-CLIPV, z));
                float ez = __expf(z);                          // tanh(z/2) = (e^z-1)/(e^z+1)
                float o  = (ez - 1.0f) * __builtin_amdgcn_rcpf(ez + 1.0f);
                // cr = rc^{-1}  =>  even input[rc[i]] = odd output[i]: scatter via rc
                bufB[slab + rc_s[((vbase + mb + r) << 4) + chan]] = (_Float16)o;
            }
        }
        __syncthreads();

        // ---- even layer: leave-one-out product + log-ratio; branch-free over 2048 rows ----
        for (unsigned g = tid; g < ROWSP; g += bs) {
            const v16h row = *(const v16h*)(bufB + (g << 4));
            float ev16[16], pre[16], suf[16];
            #pragma unroll
            for (int l = 0; l < 16; ++l) {
                float val = (float)row[l];
                ev16[l] = (val == 0.0f) ? 1.0f : val;
            }
            float acc = 1.0f;
            #pragma unroll
            for (int l = 0; l < 16; ++l) { pre[l] = acc; acc *= ev16[l]; }
            acc = 1.0f;
            #pragma unroll
            for (int l = 15; l >= 0; --l) { suf[l] = acc; acc *= ev16[l]; }
            v16h o;
            #pragma unroll
            for (int l = 0; l < 16; ++l) {
                float r = pre[l] * suf[l];
                if (t == 0) r = fminf(CLIPV, fmaxf(-CLIPV, r));
                float arg = (1.0f + r) * __builtin_amdgcn_rcpf(1.0f - r + 1e-9f) + 1e-9f;
                o[l] = (_Float16)__logf(arg);
            }
            *(v16h*)(bufA + (g << 4)) = o;
        }
        __syncthreads();
    }

    // ---- final: eo[p,v] = dot(ev[rc-row], w_e_out); reuse xv_s (stride-256 layout) ----
    for (unsigned g = tid; g < ROWSP; g += bs) {
        const unsigned slab = (g >> 8) << 12;
        const int* rcb = rc_s + ((g & 255u) << 4);
        float acc = 0.0f;
        #pragma unroll
        for (int l = 0; l < 16; ++l)
            acc += (float)bufA[slab + rcb[l]] * weo_s[l];
        xv_s[g] = acc;
    }
    __syncthreads();

    float* outr = out + (size_t)b * (NV + 1);
    for (unsigned c = tid; c < (NV + 1); c += bs) {
        float acc = xr[c];
        #pragma unroll
        for (int p = 0; p < NP; ++p) {
            int id = perma[p * (NV + 1) + c];   // 0 hits the zero-pad column
            if (id > 0) acc += xv_s[(p << 8) + (id - 1)];
        }
        outr[c] = acc;
    }
}

extern "C" void kernel_launch(void* const* d_in, const int* in_sizes, int n_in,
                              void* d_out, int out_size, void* d_ws, size_t ws_size,
                              hipStream_t stream) {
    const float* x        = (const float*)d_in[0];
    const float* message  = (const float*)d_in[1];
    const float* oddw_v   = (const float*)d_in[2];
    const float* oddw_e   = (const float*)d_in[3];
    const float* w_e_out  = (const float*)d_in[4];
    const int*   perma    = (const int*)d_in[5];
    const int*   rc_idx   = (const int*)d_in[6];
    // d_in[7] (cr_idx) unused: cr = rc^{-1}, handled as a scatter through rc.
    const int B = in_sizes[0] / (NV + 1);

    cycnet_bp_kernel<<<B, 512, 0, stream>>>(x, message, oddw_v, oddw_e,
                                            w_e_out, perma, rc_idx,
                                            (float*)d_out);
}
